// VectorQuantizer_35656818491860
// MI455X (gfx1250) — matrix operand order repaired and checked by reading, not compiled, not run on hardware
//
#include <hip/hip_runtime.h>
#include <hip/hip_bf16.h>

typedef __attribute__((ext_vector_type(16))) __bf16 v16bf;
typedef __attribute__((ext_vector_type(8)))  __bf16 v8bf;
typedef __attribute__((ext_vector_type(8)))  float  v8f;

#define EMB_DIM   128
#define NUM_CB    1024
#define N_ROWS    131072      // 8*16*32*32
#define TOTAL_EL  16777216    // 8*128*16*32*32
#define SPATIAL   16384       // 16*32*32
#define BATCH_STRIDE 2097152  // 128*16384
#define LDS_STRIDE 136        // 128 + 8 bf16 pad -> 68-dword stride, no bank conflicts
#define ROWS_PER_BLK 128
#define COLS_PER_TILE 64
#define N_TILES (NUM_CB / COLS_PER_TILE)

// ---------------------------------------------------------------------------
// Prep: convert codebook to bf16 panel + compute ||c_k||^2
// ---------------------------------------------------------------------------
__global__ void vq_prep_kernel(const float* __restrict__ cb,
                               __bf16* __restrict__ cb_bf,
                               float* __restrict__ cb_norm) {
  const int k = blockIdx.x;
  const int c = threadIdx.x;
  float v = cb[k * EMB_DIM + c];
  cb_bf[k * EMB_DIM + c] = (__bf16)v;
  __shared__ float red[EMB_DIM];
  red[c] = v * v;
  __syncthreads();
  for (int s = EMB_DIM / 2; s > 0; s >>= 1) {
    if (c < s) red[c] += red[c + s];
    __syncthreads();
  }
  if (c == 0) cb_norm[k] = red[0];
}

// ---------------------------------------------------------------------------
// Main: block = 256 threads (8 waves), 128 rows. Codebook swept in 64-column
// tiles, double-buffered in LDS via global_load_async_to_lds_b128 so tile t+1
// streams in while tile t is consumed by 16 WMMAs per wave.
// grid = 1024 blocks
// ---------------------------------------------------------------------------
__global__ void __launch_bounds__(256)
vq_main_kernel(const float* __restrict__ in,
               const float* __restrict__ cb,
               const __bf16* __restrict__ cb_bf,
               const float* __restrict__ cb_norm,
               float* __restrict__ out_q,
               float* __restrict__ out_idx) {
  __shared__ __bf16 smA[ROWS_PER_BLK * LDS_STRIDE];        // 34816 B
  __shared__ __bf16 smB[2][COLS_PER_TILE * LDS_STRIDE];    // 2 x 17408 B
  __shared__ int    idx_sm[ROWS_PER_BLK];

  const int tid   = threadIdx.x;
  const long base = (long)blockIdx.x * ROWS_PER_BLK;

  // ---- async-issue helper: 64 cols x 256 B = 1024 x 16 B = 256 thr x 4 ---
  auto issue_tile = [&](int tile, int buf) {
#pragma unroll
    for (int j = 0; j < 4; ++j) {
      int flat = tid + j * 256;
      int col  = flat >> 4;        // 0..63
      int seg  = flat & 15;        // 16 B segment within a 256 B column
      unsigned long long gaddr = (unsigned long long)(const void*)
          (cb_bf + (((long)(tile * COLS_PER_TILE + col)) << 7) + seg * 8);
      unsigned ldsa = (unsigned)(unsigned long long)(const void*)
          (&smB[buf][col * LDS_STRIDE + seg * 8]);
      asm volatile("global_load_async_to_lds_b128 %0, %1, off"
                   :: "v"(ldsa), "v"(gaddr) : "memory");
    }
  };

  // kick off tile 0 before the (long) A staging so it lands early
  issue_tile(0, 0);

  // ---- stage 128 rows of input into LDS as bf16 (coalesced over rows) ----
  {
    const int  b  = (int)(base >> 14);
    const int  s0 = (int)(base & (SPATIAL - 1));
    const float* inb = in + (long)b * BATCH_STRIDE + s0;
    const int r     = tid & 127;             // consecutive rows -> contiguous
    const int chalf = tid >> 7;              // 0 or 1
    for (int i = 0; i < 64; ++i) {
      int c = i * 2 + chalf;
      float v = inb[(long)c * SPATIAL + r];
      smA[r * LDS_STRIDE + c] = (__bf16)v;
    }
  }

  const int wave = tid >> 5;
  const int lane = tid & 31;
  const int half = lane >> 4;      // K-half selector (A/B layouts)
  const int ln   = lane & 15;      // A: row within tile; B/D: column
  const int wrow = wave * 16;

  float bestD[8];
  int   bestI[8];
#pragma unroll
  for (int v = 0; v < 8; ++v) { bestD[v] = 3.0e38f; bestI[v] = 0x7fffffff; }

  const __bf16* arow = &smA[(wrow + ln) * LDS_STRIDE];

  for (int tile = 0; tile < N_TILES; ++tile) {
    // retire this tile's async loads (issued last iteration / pre-loop)
    asm volatile("s_wait_asynccnt 0x0" ::: "memory");
    __syncthreads();   // all waves: tile data visible AND tile-1's buffer free

    // stream next tile into the alternate buffer while we compute
    if (tile + 1 < N_TILES) issue_tile(tile + 1, (tile + 1) & 1);

    const __bf16* bbase = &smB[tile & 1][0];

    // ---- compute: each wave, 4 chunks of 16 columns ----------------------
#pragma unroll
    for (int c2 = 0; c2 < 4; ++c2) {
      const int colLocal = c2 * 16 + ln;
      const int col      = tile * COLS_PER_TILE + colLocal;
      v8f acc = {};
#pragma unroll
      for (int ks = 0; ks < 4; ++ks) {
        // A fragment: lane(half) holds K = ks*32 + 8*half + {0..7, 16..23}
        const __bf16* ap = arow + ks * 32 + 8 * half;
        v8bf alo = *(const v8bf*)(ap);
        v8bf ahi = *(const v8bf*)(ap + 16);
        v16bf a = __builtin_shufflevector(alo, ahi,
                                          0, 1, 2, 3, 4, 5, 6, 7,
                                          8, 9, 10, 11, 12, 13, 14, 15);
        // B fragment: lane = column, K = ks*32 + 8*half + {0..7, 16..23}
        const __bf16* bp = bbase + colLocal * LDS_STRIDE + ks * 32 + 8 * half;
        v8bf blo = *(const v8bf*)(bp);
        v8bf bhi = *(const v8bf*)(bp + 16);
        v16bf bm = __builtin_shufflevector(blo, bhi,
                                           0, 1, 2, 3, 4, 5, 6, 7,
                                           8, 9, 10, 11, 12, 13, 14, 15);
        acc = __builtin_amdgcn_wmma_f32_16x16x32_bf16(
            false, a, false, bm, (short)0, acc, false, false);
      }
      const float cn = cb_norm[col];
#pragma unroll
      for (int v = 0; v < 8; ++v) {
        float d = __builtin_fmaf(-2.0f, acc[v], cn);
        if (d < bestD[v] || (d == bestD[v] && col < bestI[v])) {
          bestD[v] = d; bestI[v] = col;
        }
      }
    }
  }

  // ---- cross-lane argmin: 16 lanes of the same half share rows M=8*half+v
#pragma unroll
  for (int v = 0; v < 8; ++v) {
    float d = bestD[v];
    int   id = bestI[v];
    for (int m = 1; m < 16; m <<= 1) {
      float od = __shfl_xor(d, m, 32);
      int   oi = __shfl_xor(id, m, 32);
      if (od < d || (od == d && oi < id)) { d = od; id = oi; }
    }
    if (ln == 0) idx_sm[wrow + 8 * half + v] = id;
  }
  __syncthreads();

  // ---- epilogue: gather exact f32 codebook rows (coalesced over channels)
  {
    const int c  = tid & 127;
    const int r0 = tid >> 7;
    for (int i = 0; i < 64; ++i) {
      const int r = i * 2 + r0;
      const int k = idx_sm[r];
      out_q[(base + r) * EMB_DIM + c] = cb[(long)k * EMB_DIM + c];
    }
  }
  if (tid < 128) out_idx[base + tid] = (float)idx_sm[tid];
}

// ---------------------------------------------------------------------------
// Loss: 1.25 * mean((q_flat - in_flat)^2)  (faithful to reference's layout bug)
// ---------------------------------------------------------------------------
__global__ void vq_zero_loss_kernel(float* __restrict__ loss) { *loss = 0.0f; }

__global__ void vq_loss_kernel(const float* __restrict__ in,
                               const float* __restrict__ q,
                               float* __restrict__ loss) {
  __shared__ float red[256];
  const int tid = threadIdx.x;
  long i = (long)blockIdx.x * 256 + tid;
  const long stride = (long)gridDim.x * 256;
  float s = 0.0f;
  for (; i < TOTAL_EL; i += stride) {
    float d = q[i] - in[i];
    s = __builtin_fmaf(d, d, s);
  }
  red[tid] = s;
  __syncthreads();
  for (int w = 128; w > 0; w >>= 1) {
    if (tid < w) red[tid] += red[tid + w];
    __syncthreads();
  }
  if (tid == 0) {
    atomicAdd(loss, red[0] * (1.25f / (float)TOTAL_EL));
  }
}

// ---------------------------------------------------------------------------
extern "C" void kernel_launch(void* const* d_in, const int* in_sizes, int n_in,
                              void* d_out, int out_size, void* d_ws, size_t ws_size,
                              hipStream_t stream) {
  const float* in = (const float*)d_in[0];   // [8,128,16,32,32] f32
  const float* cb = (const float*)d_in[1];   // [1024,128] f32

  float* out     = (float*)d_out;
  float* out_q   = out;                                // [N*C] quantized_st
  float* out_ls  = out + (long)N_ROWS * EMB_DIM;       // [1] loss
  float* out_idx = out_ls + 1;                         // [N] indices (as float)

  __bf16* cb_bf   = (__bf16*)d_ws;                     // 256 KB
  float*  cb_norm = (float*)((char*)d_ws + NUM_CB * EMB_DIM * sizeof(__bf16));

  vq_prep_kernel<<<NUM_CB, EMB_DIM, 0, stream>>>(cb, cb_bf, cb_norm);
  vq_main_kernel<<<N_ROWS / ROWS_PER_BLK, 256, 0, stream>>>(in, cb, cb_bf, cb_norm,
                                                            out_q, out_idx);
  vq_zero_loss_kernel<<<1, 1, 0, stream>>>(out_ls);
  vq_loss_kernel<<<2048, 256, 0, stream>>>(in, out_q, out_ls);
}
